// GaussianActor_48868137894248
// MI455X (gfx1250) — matrix operand order repaired and checked by reading, not compiled, not run on hardware
//
#include <hip/hip_runtime.h>
#include <hip/hip_bf16.h>

typedef __attribute__((ext_vector_type(16))) __bf16 v16bf;
typedef __attribute__((ext_vector_type(8)))  __bf16 v8bf;
typedef __attribute__((ext_vector_type(4)))  __bf16 v4bf;
typedef __attribute__((ext_vector_type(8)))  float  v8f;

#define ROWS_PER_WG 128
#define HDIM 256
#define D_IN 48
#define D_OUT 12
#define LN_EPS 1e-5f

// ---- manual LDS layout (bytes): hot/small regions low so ds immediates fold ----
#define BIAS_OFF 0                    // float[256] transposed  (1 KB)
#define G_OFF    1024                 // float[256] transposed  (1 KB)
#define BE_OFF   2048                 // float[256] transposed  (1 KB)
#define X_OFF    3072                 // bf16[128*64]           (16 KB)
#define ACT_OFF  (X_OFF + 16384)      // bf16[128*256]          (64 KB) -> 19456
#define W_OFF    (ACT_OFF + 65536)    // bf16[256*256] W^T      (128 KB) -> 84992
#define LDS_TOTAL (W_OFF + 131072)    // 216064 B (< 320 KB WGP limit)

// ---- workspace layout (bytes): pre-converted bf16 transposed weights ----
#define WS_W1 0                       // [256][64]  32768 B
#define WS_W2 32768                   // [256][256] 131072 B
#define WS_W3 163840                  // [256][256] 131072 B
#define WS_WM 294912                  // [16][256]  8192 B
#define WS_TOTAL 303104

union V16U { v16bf v; v8bf h[2]; };

// A-fragment: 16x32 bf16 tile, contiguous k order (same as B).
// lane<16: K = k0..k0+15; lane>=16: K = k0+16..k0+31 (contiguous, matches B).
__device__ __forceinline__ v16bf load_a_frag(const __bf16* A, int row, int stride,
                                             int k0, int hh) {
  return *(const v16bf*)(A + row * stride + k0 + hh * 16);
}

// B-fragment: 32x16 bf16 from transposed weights WT[N][K]; 16 contiguous K per lane.
__device__ __forceinline__ v16bf load_b_frag(const __bf16* WT, int wk, int col,
                                             int k0, int hh) {
  return *(const v16bf*)(WT + col * wk + k0 + hh * 16);
}

template <int KSTEPS, int NT>
__device__ __forceinline__ void wave_gemm(const __bf16* A, int astride,
                                          const __bf16* WT, int lrow0, v8f acc[NT]) {
  const int lane = threadIdx.x & 31;
  const int hh = lane >> 4;
  const int lm = lane & 15;
  const int wk = KSTEPS * 32;
#pragma unroll
  for (int t = 0; t < NT; ++t)
#pragma unroll
    for (int e = 0; e < 8; ++e) acc[t][e] = 0.0f;

#pragma unroll
  for (int ks = 0; ks < KSTEPS; ++ks) {
    v16bf a = load_a_frag(A, lrow0 + lm, astride, ks * 32, hh);
#pragma unroll
    for (int t = 0; t < NT; ++t) {
      v16bf b = load_b_frag(WT, wk, t * 16 + lm, ks * 32, hh);
      acc[t] = __builtin_amdgcn_wmma_f32_16x16x32_bf16(
          false, a, false, b, (short)0, acc[t], false, false);
    }
  }
}

// bias (in-place) + LayerNorm + ReLU epilogue on one wave's 16x256 block.
// acc[t][r]: row = lrow0 + hh*8 + r, col = t*16 + lm.
// LN params are LDS-transposed: param[col] lives at pT[lm*16 + t] -> 4 float4 loads/lane.
__device__ __forceinline__ void ln_relu_store(v8f acc[16], const float* sbT,
                                              const float* sgT, const float* sbeT,
                                              __bf16* act, int lrow0) {
  const int lane = threadIdx.x & 31;
  const int hh = lane >> 4;
  const int lm = lane & 15;
  float bv[16], gv[16], bev[16];
#pragma unroll
  for (int j = 0; j < 4; ++j) {
    float4 fb = ((const float4*)(sbT + lm * 16))[j];
    float4 fg = ((const float4*)(sgT + lm * 16))[j];
    float4 fe = ((const float4*)(sbeT + lm * 16))[j];
    bv[4*j+0]=fb.x; bv[4*j+1]=fb.y; bv[4*j+2]=fb.z; bv[4*j+3]=fb.w;
    gv[4*j+0]=fg.x; gv[4*j+1]=fg.y; gv[4*j+2]=fg.z; gv[4*j+3]=fg.w;
    bev[4*j+0]=fe.x; bev[4*j+1]=fe.y; bev[4*j+2]=fe.z; bev[4*j+3]=fe.w;
  }
  // bias in place (single add per element)
#pragma unroll
  for (int t = 0; t < 16; ++t)
#pragma unroll
    for (int e = 0; e < 8; ++e) acc[t][e] += bv[t];

#pragma unroll
  for (int r = 0; r < 8; ++r) {
    float s = 0.0f, sq = 0.0f;
#pragma unroll
    for (int t = 0; t < 16; ++t) {
      float v = acc[t][r];
      s += v; sq += v * v;
    }
#pragma unroll
    for (int m = 1; m < 16; m <<= 1) {  // reduce within each 16-lane half
      s  += __shfl_xor(s,  m, 32);
      sq += __shfl_xor(sq, m, 32);
    }
    float mu  = s * (1.0f / 256.0f);
    float var = sq * (1.0f / 256.0f) - mu * mu;
    float rs  = rsqrtf(var + LN_EPS);
    int row = lrow0 + hh * 8 + r;
#pragma unroll
    for (int t = 0; t < 16; ++t) {
      float hn = (acc[t][r] - mu) * rs * gv[t] + bev[t];
      hn = fmaxf(hn, 0.0f);
      act[row * HDIM + t * 16 + lm] = (__bf16)hn;
    }
  }
}

// stage transposed LN params: pT[lm*16 + t] = p[t*16 + lm]
__device__ __forceinline__ void stage_params(float* sbT, float* sgT, float* sbeT,
                                             const float* b, const float* g,
                                             const float* be, int tid) {
  if (tid < HDIM) {
    int t = tid >> 4, lm = tid & 15;
    int d = lm * 16 + t;
    sbT[d] = b[tid]; sgT[d] = g[tid]; sbeT[d] = be[tid];
  }
}

// ---- one-shot prep: f32 weights -> bf16 transposed (padded) in workspace ----
__global__ __launch_bounds__(256) void weight_prep_kernel(
    const float* __restrict__ W1, const float* __restrict__ W2,
    const float* __restrict__ W3, const float* __restrict__ Wm,
    __bf16* __restrict__ ws) {
  int idx = blockIdx.x * 256 + threadIdx.x;   // 592*256 = 151552 total
  if (idx < 16384) {                          // W1^T [256][64], pad K 48->64
    int n = idx >> 6, k = idx & 63;
    ws[WS_W1 / 2 + idx] = (__bf16)((k < D_IN) ? W1[k * HDIM + n] : 0.0f);
  } else if (idx < 16384 + 65536) {           // W2^T [256][256]
    int i = idx - 16384; int n = i >> 8, k = i & 255;
    ws[WS_W2 / 2 + i] = (__bf16)W2[k * HDIM + n];
  } else if (idx < 16384 + 131072) {          // W3^T [256][256]
    int i = idx - (16384 + 65536); int n = i >> 8, k = i & 255;
    ws[WS_W3 / 2 + i] = (__bf16)W3[k * HDIM + n];
  } else if (idx < 16384 + 131072 + 4096) {   // Wm^T [16][256], pad N 12->16
    int i = idx - (16384 + 131072); int n = i >> 8, k = i & 255;
    ws[WS_WM / 2 + i] = (__bf16)((n < D_OUT) ? Wm[k * D_OUT + n] : 0.0f);
  }
}

template <bool PREPPED>
__global__ __launch_bounds__(256) void mlp_fused_wmma_kernel(
    const float* __restrict__ x,
    const float* __restrict__ W1, const float* __restrict__ b1,
    const float* __restrict__ g1, const float* __restrict__ be1,
    const float* __restrict__ W2, const float* __restrict__ b2,
    const float* __restrict__ g2, const float* __restrict__ be2,
    const float* __restrict__ W3, const float* __restrict__ b3,
    const float* __restrict__ g3, const float* __restrict__ be3,
    const float* __restrict__ Wm, const float* __restrict__ bm,
    const __bf16* __restrict__ wpre,
    float* __restrict__ out) {
  __shared__ __align__(16) unsigned char s_mem[LDS_TOTAL];
  float*  sbT  = (float*)(s_mem + BIAS_OFF);
  float*  sgT  = (float*)(s_mem + G_OFF);
  float*  sbeT = (float*)(s_mem + BE_OFF);
  __bf16* s_x  = (__bf16*)(s_mem + X_OFF);
  __bf16* s_act = (__bf16*)(s_mem + ACT_OFF);
  __bf16* s_w  = (__bf16*)(s_mem + W_OFF);

  const int tid = threadIdx.x;
  const int wave = tid >> 5;
  const int lrow0 = wave * 16;
  const long grow0 = (long)blockIdx.x * ROWS_PER_WG;

  // ---- stage x tile: float4 reads, packed 4x bf16 LDS writes ----
#pragma unroll
  for (int j = 0; j < 6; ++j) {                // 128*48 = 1536 float4 chunks
    int idx = tid + j * 256;
    int r = idx / 12, c = (idx % 12) * 4;
    float4 f = *(const float4*)(x + (grow0 + r) * D_IN + c);
    v4bf o = { (__bf16)f.x, (__bf16)f.y, (__bf16)f.z, (__bf16)f.w };
    *(v4bf*)(s_x + r * 64 + c) = o;
  }
#pragma unroll
  for (int j = 0; j < 2; ++j) {                // zero K pad 48..63
    int idx = tid + j * 256;
    int r = idx / 4, c = 48 + (idx % 4) * 4;
    *(v4bf*)(s_x + r * 64 + c) = (v4bf){(__bf16)0.f, (__bf16)0.f, (__bf16)0.f, (__bf16)0.f};
  }
  // ---- stage W1^T + LN params ----
  if (PREPPED) {
    const v8bf* src = (const v8bf*)(wpre + WS_W1 / 2);
    for (int i = tid; i < 2048; i += 256) ((v8bf*)s_w)[i] = src[i];
  } else {
    for (int i = tid; i < 256 * 64; i += 256) {
      int n = i >> 6, k = i & 63;
      s_w[i] = (__bf16)((k < D_IN) ? W1[k * HDIM + n] : 0.0f);
    }
  }
  stage_params(sbT, sgT, sbeT, b1, g1, be1, tid);
  __syncthreads();

  v8f acc[16];

  // ---- layer 1: 48(64)->256 ----
  wave_gemm<2, 16>(s_x, 64, s_w, lrow0, acc);
  ln_relu_store(acc, sbT, sgT, sbeT, s_act, lrow0);
  __syncthreads();

  // ---- layer 2: 256->256 ----
  if (PREPPED) {
    const v8bf* src = (const v8bf*)(wpre + WS_W2 / 2);
    for (int i = tid; i < 8192; i += 256) ((v8bf*)s_w)[i] = src[i];
  } else {
    for (int i = tid; i < HDIM * HDIM; i += 256) {
      int n = i >> 8, k = i & 255;
      s_w[i] = (__bf16)W2[k * HDIM + n];
    }
  }
  stage_params(sbT, sgT, sbeT, b2, g2, be2, tid);
  __syncthreads();
  wave_gemm<8, 16>(s_act, HDIM, s_w, lrow0, acc);
  ln_relu_store(acc, sbT, sgT, sbeT, s_act, lrow0);
  __syncthreads();

  // ---- layer 3: 256->256 ----
  if (PREPPED) {
    const v8bf* src = (const v8bf*)(wpre + WS_W3 / 2);
    for (int i = tid; i < 8192; i += 256) ((v8bf*)s_w)[i] = src[i];
  } else {
    for (int i = tid; i < HDIM * HDIM; i += 256) {
      int n = i >> 8, k = i & 255;
      s_w[i] = (__bf16)W3[k * HDIM + n];
    }
  }
  stage_params(sbT, sgT, sbeT, b3, g3, be3, tid);
  __syncthreads();
  wave_gemm<8, 16>(s_act, HDIM, s_w, lrow0, acc);
  ln_relu_store(acc, sbT, sgT, sbeT, s_act, lrow0);
  __syncthreads();

  // ---- layer 4: 256->12 (pad N to 16), tanh, store ----
  if (PREPPED) {
    const v8bf* src = (const v8bf*)(wpre + WS_WM / 2);
    for (int i = tid; i < 512; i += 256) ((v8bf*)s_w)[i] = src[i];
  } else {
    for (int i = tid; i < 16 * HDIM; i += 256) {
      int n = i >> 8, k = i & 255;
      s_w[i] = (__bf16)((n < D_OUT) ? Wm[k * D_OUT + n] : 0.0f);
    }
  }
  if (tid < 16) sbT[tid] = (tid < D_OUT) ? bm[tid] : 0.0f;
  __syncthreads();
  v8f acc4[1];
  wave_gemm<8, 1>(s_act, HDIM, s_w, lrow0, acc4);
  {
    const int lane = tid & 31;
    const int hh = lane >> 4;
    const int lm = lane & 15;
    float bmv = sbT[lm];
#pragma unroll
    for (int r = 0; r < 8; ++r) {
      float v = tanhf(acc4[0][r] + bmv);
      long row = grow0 + lrow0 + hh * 8 + r;
      if (lm < D_OUT) out[row * D_OUT + lm] = v;
    }
  }
}

extern "C" void kernel_launch(void* const* d_in, const int* in_sizes, int n_in,
                              void* d_out, int out_size, void* d_ws, size_t ws_size,
                              hipStream_t stream) {
  const float* x   = (const float*)d_in[0];
  const float* W1  = (const float*)d_in[1];
  const float* b1  = (const float*)d_in[2];
  const float* g1  = (const float*)d_in[3];
  const float* be1 = (const float*)d_in[4];
  const float* W2  = (const float*)d_in[5];
  const float* b2  = (const float*)d_in[6];
  const float* g2  = (const float*)d_in[7];
  const float* be2 = (const float*)d_in[8];
  const float* W3  = (const float*)d_in[9];
  const float* b3  = (const float*)d_in[10];
  const float* g3  = (const float*)d_in[11];
  const float* be3 = (const float*)d_in[12];
  const float* Wm  = (const float*)d_in[13];
  const float* bm  = (const float*)d_in[14];
  float* out = (float*)d_out;

  int rows = in_sizes[0] / D_IN;           // 262144
  int grid = rows / ROWS_PER_WG;           // 2048

  if (ws_size >= (size_t)WS_TOTAL) {
    __bf16* ws = (__bf16*)d_ws;
    weight_prep_kernel<<<592, 256, 0, stream>>>(W1, W2, W3, Wm, ws);
    mlp_fused_wmma_kernel<true><<<grid, 256, 0, stream>>>(
        x, W1, b1, g1, be1, W2, b2, g2, be2, W3, b3, g3, be3, Wm, bm, ws, out);
  } else {
    mlp_fused_wmma_kernel<false><<<grid, 256, 0, stream>>>(
        x, W1, b1, g1, be1, W2, b2, g2, be2, W3, b3, g3, be3, Wm, bm, nullptr, out);
  }
}
